// QuantizedAttention_73684458930244
// MI455X (gfx1250) — matrix operand order repaired and checked by reading, not compiled, not run on hardware
//
#include <hip/hip_runtime.h>
#include <hip/hip_bf16.h>

// QuantizedAttention for MI455X (gfx1250, wave32).
// quantize(e5m10, RTE, clamp) == saturating f32->f16 conversion, so all
// post-quantize GEMMs run exactly on V_WMMA_F32_16X16X32_F16 (f32 accum).
// QKV uses hi/lo split-f16 (3 WMMAs) to preserve f32 input precision.
// The f16 GEMMs use GLOBAL_LOAD_ASYNC_TO_LDS_B128 (ASYNCcnt) with
// double-buffered LDS so global->LDS traffic overlaps the WMMA stream.
// Workspace layout (needs 80 MiB):
//   [0,16M)  q   f16 [8192][1024]
//   [16,32M) k   f16 [8192][1024]
//   [32,48M) vT  f16 [4][1024][2048]   (v stored transposed for NT AV-GEMM)
//   [48,80M) scores/attn f16 [4][2048][2048]  (softmax in-place)

typedef __attribute__((ext_vector_type(16))) _Float16 v16h;
typedef __attribute__((ext_vector_type(8)))  _Float16 v8h;
typedef __attribute__((ext_vector_type(8)))  float    v8f;
typedef __attribute__((ext_vector_type(4)))  float    v4f;
typedef __attribute__((ext_vector_type(4)))  int      v4i;

#define TB 4
#define TS 2048
#define TH 1024
#define TM (TB * TS)

#define BM 128
#define BN 128
#define BK 32
#define LDST 40   // padded LDS row stride in halves (80B: bank-rotating, 16B aligned)

#ifndef __has_builtin
#define __has_builtin(x) 0
#endif
#if __has_builtin(__builtin_amdgcn_global_load_async_to_lds_b128)
#define HAVE_ASYNC_LDS 1
#else
#define HAVE_ASYNC_LDS 0
#endif

__device__ __forceinline__ v8f wmma_f16(v16h a, v16h b, v8f c) {
  // (neg_a, A, neg_b, B, c_mod, C, reuse_a, reuse_b)
  return __builtin_amdgcn_wmma_f32_16x16x32_f16(false, a, false, b, (short)0, c,
                                                false, false);
}

// A fragment (16x32 f16): lanes 0-15 row M=lane, halves 0..7 -> K 0..7,
// halves 8..15 -> K 16..23; lanes 16-31: K 8..15 / 24..31.
__device__ __forceinline__ v16h frag_a(const _Float16* tile, int m0, int lane) {
  const _Float16* p = tile + (m0 + (lane & 15)) * LDST + ((lane >> 4) << 3);
  v8h lo = *(const v8h*)p;
  v8h hi = *(const v8h*)(p + 16);
  v16h r;
#pragma unroll
  for (int i = 0; i < 8; ++i) { r[i] = lo[i]; r[i + 8] = hi[i]; }
  return r;
}

// B fragment (32x16 f16): col N = lane&15, same K order as frag_a
// (lanes 0-15: K 0..7 / 16..23; lanes 16-31: K 8..15 / 24..31).
__device__ __forceinline__ v16h frag_b(const _Float16* tile, int n0, int lane) {
  const _Float16* p = tile + (n0 + (lane & 15)) * LDST + ((lane >> 4) << 3);
  v8h lo = *(const v8h*)p;
  v8h hi = *(const v8h*)(p + 16);
  v16h r;
#pragma unroll
  for (int i = 0; i < 8; ++i) { r[i] = lo[i]; r[i + 8] = hi[i]; }
  return r;
}

#if HAVE_ASYNC_LDS
// builtin signature (from clang diagnostic): (v4i AS1* gsrc, v4i AS3* ldst,
// imm offset, imm cpol); 16 bytes per lane, tracked by ASYNCcnt.
typedef __attribute__((address_space(1))) v4i* gv4i_p;
typedef __attribute__((address_space(3))) v4i* lv4i_p;

// Issue one 128x32 A tile + one 128x32 B tile as async global->LDS copies.
// 4 x b128 per wave per tile -> wave ASYNCcnt += 4.
__device__ __forceinline__ void async_tile(const _Float16* __restrict__ gA,
                                           const _Float16* __restrict__ gB,
                                           _Float16* sA, _Float16* sB,
                                           int tid, int m0, int n0, int K, int k0) {
#pragma unroll
  for (int i = 0; i < 2; ++i) {
    int slot = tid + (i << 8);         // 0..511, 16B each
    int r = slot >> 2;                 // 0..127
    int c = (slot & 3) << 3;           // 0,8,16,24 halves
    __builtin_amdgcn_global_load_async_to_lds_b128(
        (gv4i_p)(gA + (long)(m0 + r) * K + k0 + c),
        (lv4i_p)(sA + r * LDST + c), 0, 0);
    __builtin_amdgcn_global_load_async_to_lds_b128(
        (gv4i_p)(gB + (long)(n0 + r) * K + k0 + c),
        (lv4i_p)(sB + r * LDST + c), 0, 0);
  }
}
#endif

// ---------------------------------------------------------------------------
// Kernel 1: y = f16sat(X @ W^T + b). X:[8192,1024]f32, W:[1024,1024]f32.
// Split-f16: X=xh+xl, W=wh+wl; acc += xh*wh + xh*wl + xl*wh (err ~2^-22).
// storeT: store output transposed per batch as [b][o][s] (used for V).
// (Keeps the VGPR-staged path: the f32 -> hi/lo-f16 split needs VALU.)
// ---------------------------------------------------------------------------
__global__ void __launch_bounds__(256)
qkv_gemm(const float* __restrict__ X, const float* __restrict__ W,
         const float* __restrict__ bias, _Float16* __restrict__ Out, int storeT) {
  __shared__ __align__(16) _Float16 lAh[BM * LDST];
  __shared__ __align__(16) _Float16 lAl[BM * LDST];
  __shared__ __align__(16) _Float16 lBh[BN * LDST];
  __shared__ __align__(16) _Float16 lBl[BN * LDST];

  const int tid = threadIdx.x;
  const int lane = tid & 31, wave = tid >> 5;
  const int wm = (wave >> 2) * 64, wn = (wave & 3) * 32;  // 2x4 wave grid
  const int m0 = blockIdx.x * BM, n0 = blockIdx.y * BN;

  v8f zero = {};
  v8f acc[4][2];
#pragma unroll
  for (int mi = 0; mi < 4; ++mi)
#pragma unroll
    for (int ni = 0; ni < 2; ++ni) acc[mi][ni] = zero;

  for (int k0 = 0; k0 < TH; k0 += BK) {
    // cooperative f32 load + hi/lo split into LDS (16 floats per thread)
#pragma unroll
    for (int i = 0; i < 4; ++i) {
      int slot = tid + (i << 8);         // 0..1023, 4 floats each
      int r = slot >> 3;                 // 0..127
      int c = (slot & 7) << 2;           // 0,4,...,28
      v4f xa = *(const v4f*)&X[(long)(m0 + r) * TH + k0 + c];
      v4f xb = *(const v4f*)&W[(long)(n0 + r) * TH + k0 + c];
#pragma unroll
      for (int j = 0; j < 4; ++j) {
        _Float16 ha = (_Float16)xa[j];
        lAh[r * LDST + c + j] = ha;
        lAl[r * LDST + c + j] = (_Float16)(xa[j] - (float)ha);
        _Float16 hb = (_Float16)xb[j];
        lBh[r * LDST + c + j] = hb;
        lBl[r * LDST + c + j] = (_Float16)(xb[j] - (float)hb);
      }
    }
    if (k0 + BK < TH) {  // speculative L2 prefetch of next X tile rows
      __builtin_prefetch(&X[(long)(m0 + (tid >> 1)) * TH + k0 + BK], 0, 0);
    }
    __syncthreads();

    v16h bh[2], bl[2];
#pragma unroll
    for (int ni = 0; ni < 2; ++ni) {
      bh[ni] = frag_b(lBh, wn + ni * 16, lane);
      bl[ni] = frag_b(lBl, wn + ni * 16, lane);
    }
#pragma unroll
    for (int mi = 0; mi < 4; ++mi) {
      v16h ah = frag_a(lAh, wm + mi * 16, lane);
      v16h al = frag_a(lAl, wm + mi * 16, lane);
#pragma unroll
      for (int ni = 0; ni < 2; ++ni) {
        acc[mi][ni] = wmma_f16(ah, bh[ni], acc[mi][ni]);
        acc[mi][ni] = wmma_f16(ah, bl[ni], acc[mi][ni]);
        acc[mi][ni] = wmma_f16(al, bh[ni], acc[mi][ni]);
      }
    }
    __syncthreads();
  }

  // epilogue: +bias, saturating quantize to f16 (== reference quantize)
#pragma unroll
  for (int mi = 0; mi < 4; ++mi) {
#pragma unroll
    for (int ni = 0; ni < 2; ++ni) {
      int gn = n0 + wn + ni * 16 + (lane & 15);
      int gmBase = m0 + wm + mi * 16 + ((lane >> 4) << 3);
      float bval = bias[gn];
#pragma unroll
      for (int r = 0; r < 8; ++r) {
        float y = acc[mi][ni][r] + bval;
        y = fminf(fmaxf(y, -65504.0f), 65504.0f);
        _Float16 qv = (_Float16)y;
        int gm = gmBase + r;
        if (storeT) {
          int b = gm >> 11;          // / TS
          int s = gm & (TS - 1);
          Out[((long)(b * TH + gn) << 11) + s] = qv;   // [b][o][s]
        } else {
          Out[(long)gm * TH + gn] = qv;                // [m][o]
        }
      }
    }
  }
}

// ---------------------------------------------------------------------------
// Batched NT GEMM on f16 operands: C = scale * (A @ B^T).
// Double-buffered LDS fed by async global->LDS copies (ASYNCcnt) when
// available; falls back to VGPR-staged copies otherwise.
// F32OUT=false -> saturating f16 store (quantize); true -> f32 store.
// ---------------------------------------------------------------------------
template <bool F32OUT>
__global__ void __launch_bounds__(256)
gemm_nt(const _Float16* __restrict__ A, const _Float16* __restrict__ Bm,
        void* __restrict__ Cout, float scale, int K, int N,
        long batA, long batB, long batC) {
  __shared__ __align__(16) _Float16 lA[2][BM * LDST];
  __shared__ __align__(16) _Float16 lB[2][BN * LDST];

  const int tid = threadIdx.x;
  const int lane = tid & 31, wave = tid >> 5;
  const int wm = (wave >> 2) * 64, wn = (wave & 3) * 32;
  const int m0 = blockIdx.x * BM, n0 = blockIdx.y * BN;
  const _Float16* Ab = A + (long)blockIdx.z * batA;
  const _Float16* Bb = Bm + (long)blockIdx.z * batB;

  v8f zero = {};
  v8f acc[4][2];
#pragma unroll
  for (int mi = 0; mi < 4; ++mi)
#pragma unroll
    for (int ni = 0; ni < 2; ++ni) acc[mi][ni] = zero;

#if HAVE_ASYNC_LDS
  const int ntiles = K / BK;
  async_tile(Ab, Bb, lA[0], lB[0], tid, m0, n0, K, 0);
  for (int t = 0; t < ntiles; ++t) {
    const int cur = t & 1;
    if (t + 1 < ntiles) {
      // issue next tile into the other buffer (its readers synced at the
      // trailing barrier of iteration t-1), then wait for the current tile:
      // 4 newest asyncs may remain outstanding.
      async_tile(Ab, Bb, lA[cur ^ 1], lB[cur ^ 1], tid, m0, n0, K, (t + 1) * BK);
      asm volatile("s_wait_asynccnt 0x4" ::: "memory");
    } else {
      asm volatile("s_wait_asynccnt 0x0" ::: "memory");
    }
    __syncthreads();

    v16h fb[2];
#pragma unroll
    for (int ni = 0; ni < 2; ++ni) fb[ni] = frag_b(lB[cur], wn + ni * 16, lane);
#pragma unroll
    for (int mi = 0; mi < 4; ++mi) {
      v16h fa = frag_a(lA[cur], wm + mi * 16, lane);
#pragma unroll
      for (int ni = 0; ni < 2; ++ni)
        acc[mi][ni] = wmma_f16(fa, fb[ni], acc[mi][ni]);
    }
    __syncthreads();
  }
#else
  for (int k0 = 0; k0 < K; k0 += BK) {
#pragma unroll
    for (int i = 0; i < 2; ++i) {
      int slot = tid + (i << 8);         // 0..511, 8 halves each
      int r = slot >> 2;                 // 0..127
      int c = (slot & 3) << 3;           // 0,8,16,24
      *(v8h*)&lA[0][r * LDST + c] = *(const v8h*)&Ab[(long)(m0 + r) * K + k0 + c];
      *(v8h*)&lB[0][r * LDST + c] = *(const v8h*)&Bb[(long)(n0 + r) * K + k0 + c];
    }
    __syncthreads();
    v16h fb[2];
#pragma unroll
    for (int ni = 0; ni < 2; ++ni) fb[ni] = frag_b(lB[0], wn + ni * 16, lane);
#pragma unroll
    for (int mi = 0; mi < 4; ++mi) {
      v16h fa = frag_a(lA[0], wm + mi * 16, lane);
#pragma unroll
      for (int ni = 0; ni < 2; ++ni)
        acc[mi][ni] = wmma_f16(fa, fb[ni], acc[mi][ni]);
    }
    __syncthreads();
  }
#endif

  long cb = (long)blockIdx.z * batC;
#pragma unroll
  for (int mi = 0; mi < 4; ++mi) {
#pragma unroll
    for (int ni = 0; ni < 2; ++ni) {
      int gn = n0 + wn + ni * 16 + (lane & 15);
      int gmBase = m0 + wm + mi * 16 + ((lane >> 4) << 3);
#pragma unroll
      for (int r = 0; r < 8; ++r) {
        float y = acc[mi][ni][r] * scale;
        long idx = cb + (long)(gmBase + r) * N + gn;
        if constexpr (F32OUT) {
          ((float*)Cout)[idx] = y;
        } else {
          y = fminf(fmaxf(y, -65504.0f), 65504.0f);
          ((_Float16*)Cout)[idx] = (_Float16)y;
        }
      }
    }
  }
}

// ---------------------------------------------------------------------------
// Row softmax over 2048 f16 scores, f32 math, in-place f16 (quantized) out.
// One 256-thread workgroup (8 waves) per row; 8 elements per thread.
// ---------------------------------------------------------------------------
__global__ void __launch_bounds__(256)
softmax_rows(_Float16* __restrict__ P) {
  __shared__ float red[256];
  const int t = threadIdx.x;
  _Float16* row = P + ((long)blockIdx.x << 11);  // * 2048

  float x[8];
  float m = -3.4e38f;
#pragma unroll
  for (int i = 0; i < 8; ++i) {
    x[i] = (float)row[t + (i << 8)];
    m = fmaxf(m, x[i]);
  }
  red[t] = m;
  __syncthreads();
  for (int s2 = 128; s2 > 0; s2 >>= 1) {
    if (t < s2) red[t] = fmaxf(red[t], red[t + s2]);
    __syncthreads();
  }
  m = red[0];
  __syncthreads();

  float sum = 0.0f;
#pragma unroll
  for (int i = 0; i < 8; ++i) {
    x[i] = expf(x[i] - m);
    sum += x[i];
  }
  red[t] = sum;
  __syncthreads();
  for (int s2 = 128; s2 > 0; s2 >>= 1) {
    if (t < s2) red[t] += red[t + s2];
    __syncthreads();
  }
  float inv = 1.0f / red[0];
#pragma unroll
  for (int i = 0; i < 8; ++i) row[t + (i << 8)] = (_Float16)(x[i] * inv);
}

// ---------------------------------------------------------------------------
extern "C" void kernel_launch(void* const* d_in, const int* in_sizes, int n_in,
                              void* d_out, int out_size, void* d_ws, size_t ws_size,
                              hipStream_t stream) {
  const float* x  = (const float*)d_in[0];
  const float* Wq = (const float*)d_in[1];
  const float* bq = (const float*)d_in[2];
  const float* Wk = (const float*)d_in[3];
  const float* bk = (const float*)d_in[4];
  const float* Wv = (const float*)d_in[5];
  const float* bv = (const float*)d_in[6];
  float* out = (float*)d_out;

  char* ws = (char*)d_ws;
  _Float16* q  = (_Float16*)ws;                          // 16 MiB
  _Float16* kq = (_Float16*)(ws + ((size_t)16 << 20));   // 16 MiB
  _Float16* vT = (_Float16*)(ws + ((size_t)32 << 20));   // 16 MiB
  _Float16* sc = (_Float16*)(ws + ((size_t)48 << 20));   // 32 MiB

  dim3 blk(256);

  // q/k/v projections (split-f16 WMMA); v stored transposed per batch.
  dim3 g1(TM / BM, TH / BN);  // 64 x 8
  qkv_gemm<<<g1, blk, 0, stream>>>(x, Wq, bq, q, 0);
  qkv_gemm<<<g1, blk, 0, stream>>>(x, Wk, bk, kq, 0);
  qkv_gemm<<<g1, blk, 0, stream>>>(x, Wv, bv, vT, 1);

  // scores = f16(q @ k^T / 32), per batch
  dim3 g2(TS / BM, TS / BN, TB);  // 16 x 16 x 4
  gemm_nt<false><<<g2, blk, 0, stream>>>(q, kq, sc, 0.03125f, TH, TS,
                                         (long)TS * TH, (long)TS * TH,
                                         (long)TS * TS);

  // attn = f16(softmax(scores)), in place
  softmax_rows<<<dim3(TB * TS), blk, 0, stream>>>(sc);

  // out = attn @ v  (B operand is vT so this is again an NT GEMM), f32 out
  dim3 g3(TS / BM, TH / BN, TB);  // 16 x 8 x 4
  gemm_nt<true><<<g3, blk, 0, stream>>>(sc, vT, out, 1.0f, TS, TH,
                                        (long)TS * TS, (long)TH * TS,
                                        (long)TS * TH);
}
